// VectorQuantizerEMA_16647293239450
// MI455X (gfx1250) — matrix operand order repaired and checked by reading, not compiled, not run on hardware
//
#include <hip/hip_runtime.h>
#include <hip/hip_bf16.h>
#include <math.h>
#include <stdint.h>

// Problem constants (from reference setup_inputs)
// B=16, T=512, H=8, K=1024, C=256 ; N = B*T = 8192
#define VQ_N 8192
#define VQ_H 8
#define VQ_K 1024
#define VQ_C 256
#define VQ_HC 2048            // H*C
#define VQ_NH 65536           // N*H
#define VQ_QELEMS 16777216    // N*H*C  (quantized element count)

typedef _Float16 half16  __attribute__((ext_vector_type(16)));
typedef float    float8  __attribute__((ext_vector_type(8)));
typedef float    floatx2 __attribute__((ext_vector_type(2)));   // clang vector (NT-store capable)

#define CHUNK 64              // codes resident in LDS per buffer
#define LDS_ROW 264           // 256 f16 + 8 f16 pad -> conflict-free ds_load_b128
#define NCHUNKS (VQ_K / CHUNK)

// ---------------------------------------------------------------------------
// Kernel 1: convert embedding fp32 -> f16 and compute per-code squared norms.
// grid = H*K blocks (8192), block = 256 threads (one dim each).
// ---------------------------------------------------------------------------
__global__ __launch_bounds__(256) void vq_prep(const float* __restrict__ emb,
                                               _Float16* __restrict__ embF16,
                                               float* __restrict__ normE) {
  const int code = blockIdx.x;          // flat (h*K + k)
  const int t = threadIdx.x;
  const float v = emb[(size_t)code * VQ_C + t];
  embF16[(size_t)code * VQ_C + t] = (_Float16)v;

  __shared__ float red[256];
  red[t] = v * v;
  __syncthreads();
  for (int s = 128; s > 0; s >>= 1) {
    if (t < s) red[t] += red[t + s];
    __syncthreads();
  }
  if (t == 0) normE[code] = red[0];
}

// ---------------------------------------------------------------------------
// Kernel 2: distance GEMM + argmin via WMMA, with double-buffered
// GLOBAL_LOAD_ASYNC_TO_LDS_B128 pipelining (ASYNCcnt) of the code chunks.
// grid = (N/128, H), block = 256 threads = 8 waves; each wave owns 16 rows.
// Score(n,k) = ||e_k||^2 - 2 * <x_n, e_k>   (||x||^2 is row-constant).
// ---------------------------------------------------------------------------
__global__ __launch_bounds__(256) void vq_argmin(const float* __restrict__ x,
                                                 const _Float16* __restrict__ embF16,
                                                 const float* __restrict__ normE,
                                                 int* __restrict__ idxOut) {
  __shared__ _Float16 ldsE[2][CHUNK * LDS_ROW];  // 2 x 33 KB double buffer
  __shared__ float    ldsN[VQ_K];                // all 1024 norms for this head

  const int h = blockIdx.y;
  const int rowBase = blockIdx.x * 128;
  const int tid = threadIdx.x;
  const int lane = tid & 31;
  const int wave = tid >> 5;
  const int halfSel = lane >> 4;     // 0: lanes 0-15, 1: lanes 16-31
  const int l15 = lane & 15;

  const _Float16* embHead = embF16 + (size_t)h * VQ_K * VQ_C;

  // ---- async copy of one 64x256 f16 chunk: each thread moves 128 B ----
  const int r = tid >> 2;            // 0..63 code row
  const int q = tid & 3;             // quarter of the row
  const _Float16* srcBase = embHead + r * VQ_C + q * 64;
  auto issueChunk = [&](int chunk, int buf) {
    const char* g = (const char*)(srcBase + (size_t)chunk * CHUNK * VQ_C);
    // generic shared pointer lives in the LDS aperture: addr[31:0] == LDS offset
    unsigned laddr = (unsigned)(uintptr_t)&ldsE[buf][r * LDS_ROW + q * 64];
#pragma unroll
    for (int i = 0; i < 8; ++i) {
      asm volatile("global_load_async_to_lds_b128 %0, %1, off"
                   :: "v"(laddr + 16u * i), "v"(g + 16 * i)
                   : "memory");
    }
  };

  issueChunk(0, 0);  // overlap with A-fragment conversion below

  // ---- Preload A fragments: this wave's 16 rows, all C=256 (8 k-steps) ----
  // A operand, this code's contiguous k order (matches B): lanes 0-15 row=lane, elems 0-15 -> K 0..15;
  // lanes 16-31 row=lane-16, elems 0-15 -> K 16..31.
  const int gRow = rowBase + wave * 16 + l15;
  const float* xrow = x + (size_t)gRow * VQ_HC + h * VQ_C;
  half16 afrag[8];
#pragma unroll
  for (int s = 0; s < 8; ++s) {
    const int k0 = s * 32;
    const int kb = halfSel * 16;
#pragma unroll
    for (int i = 0; i < 8; ++i) {
      afrag[s][i]     = (_Float16)xrow[k0 + kb + i];
      afrag[s][i + 8] = (_Float16)xrow[k0 + kb + 8 + i];
    }
  }

  // all norms for this head (4 KB, loaded once)
  for (int k = tid; k < VQ_K; k += 256) ldsN[k] = normE[h * VQ_K + k];

  float bestV[8];
  int   bestI[8];
#pragma unroll
  for (int j = 0; j < 8; ++j) { bestV[j] = 3.0e38f; bestI[j] = 0x7fffffff; }

  for (int chunk = 0; chunk < NCHUNKS; ++chunk) {
    // wait for this wave's async loads, then publish to all waves
    asm volatile("s_wait_asynccnt 0x0" ::: "memory");
    __syncthreads();
    // prefetch-issue the next chunk into the other buffer (WAR safe: all
    // waves finished reading it before the barrier above)
    if (chunk + 1 < NCHUNKS) issueChunk(chunk + 1, (chunk + 1) & 1);
    if (chunk + 2 < NCHUNKS)
      __builtin_prefetch(embHead + (size_t)(chunk + 2) * CHUNK * VQ_C + (size_t)tid * 64, 0, 3);

    const _Float16* eb = &ldsE[chunk & 1][0];

    // 4 sub-tiles of 16 codes each
#pragma unroll
    for (int ct = 0; ct < 4; ++ct) {
      float8 acc = {};
      // B operand, this code's contiguous k order (same as A): col = lane&15; lanes 0-15 elems -> K 0..15,
      // lanes 16-31 elems -> K 16..31 (per k-step of 32).
      const _Float16* bbase = eb + (ct * 16 + l15) * LDS_ROW + halfSel * 16;
#pragma unroll
      for (int s = 0; s < 8; ++s) {
        half16 bfrag = *(const half16*)(bbase + s * 32);
        acc = __builtin_amdgcn_wmma_f32_16x16x32_f16(
            /*neg_a=*/false, afrag[s], /*neg_b=*/false, bfrag,
            /*c_mod=*/(short)0, acc, /*reuse_a=*/false, /*reuse_b=*/false);
      }
      const int code = chunk * CHUNK + ct * 16 + l15;
      const float nE = ldsN[code];
#pragma unroll
      for (int j = 0; j < 8; ++j) {
        const float score = nE - 2.0f * acc[j];
        if (score < bestV[j]) { bestV[j] = score; bestI[j] = code; }
      }
    }
  }

  // Final per-row min across the 16 column classes (xor masks < 16 keep the
  // two row-halves of the D layout separate). Tie-break: smaller code index.
#pragma unroll
  for (int j = 0; j < 8; ++j) {
    float v = bestV[j];
    int   bi = bestI[j];
#pragma unroll
    for (int off = 1; off < 16; off <<= 1) {
      const float ov = __shfl_xor(v, off, 32);
      const int   oi = __shfl_xor(bi, off, 32);
      if (ov < v || (ov == v && oi < bi)) { v = ov; bi = oi; }
    }
    if (l15 == 0) {                       // lanes 0 (rows 0..7) and 16 (rows 8..15)
      const int m = j + halfSel * 8;
      idxOut[(size_t)(rowBase + wave * 16 + m) * VQ_H + h] = bi;
    }
  }
}

// ---------------------------------------------------------------------------
// Kernel 3: gather quantized (fp32), one-hot encodings, counts, loss partials.
// grid = (N, H), block = 256 (one dim per thread). The 335 MB of write-once
// outputs use non-temporal stores so L2 keeps x / embedding instead.
// ---------------------------------------------------------------------------
__global__ __launch_bounds__(256) void vq_scatter(const float* __restrict__ x,
                                                  const float* __restrict__ emb,
                                                  const int* __restrict__ idxBuf,
                                                  float* __restrict__ quantOut,
                                                  float* __restrict__ encOut,
                                                  float* __restrict__ counts,
                                                  float* __restrict__ lossPartial) {
  const int n = blockIdx.x;
  const int h = blockIdx.y;
  const int t = threadIdx.x;
  const int code = idxBuf[n * VQ_H + h];

  const float q  = emb[((size_t)h * VQ_K + code) * VQ_C + t];
  const float xv = x[(size_t)n * VQ_HC + h * VQ_C + t];
  // scalar NT store: base (out+1) is only 4B-aligned
  __builtin_nontemporal_store(q, &quantOut[(size_t)n * VQ_HC + h * VQ_C + t]);

  const float d = q - xv;
  __shared__ float red[256];
  red[t] = d * d;
  __syncthreads();
  for (int s = 128; s > 0; s >>= 1) {
    if (t < s) red[t] += red[t + s];
    __syncthreads();
  }
  if (t == 0) {
    lossPartial[n * VQ_H + h] = red[0];            // deterministic (no float atomic)
    atomicAdd(&counts[h * VQ_K + code], 1.0f);     // integer-valued: exact & order-free
  }

  // one-hot row: 1024 floats, 4 per thread; base is 8B-aligned -> 2x NT b64 stores
  const int kbase = t * 4;
  floatx2 lo = {0.f, 0.f};
  floatx2 hi = {0.f, 0.f};
  if (code == kbase)     lo.x = 1.0f;
  if (code == kbase + 1) lo.y = 1.0f;
  if (code == kbase + 2) hi.x = 1.0f;
  if (code == kbase + 3) hi.y = 1.0f;
  float* enc = &encOut[((size_t)n * VQ_H + h) * VQ_K + kbase];
  __builtin_nontemporal_store(lo, (floatx2*)(enc));
  __builtin_nontemporal_store(hi, (floatx2*)(enc + 2));
}

// ---------------------------------------------------------------------------
// Kernel 4: finalize loss + perplexity (single block, deterministic).
// ---------------------------------------------------------------------------
__global__ __launch_bounds__(256) void vq_finalize(const float* __restrict__ counts,
                                                   const float* __restrict__ lossPartial,
                                                   float* __restrict__ out) {
  __shared__ float red[256];
  const int t = threadIdx.x;

  float s = 0.f;
  for (int i = t; i < VQ_NH; i += 256) s += lossPartial[i];
  red[t] = s;
  __syncthreads();
  for (int w = 128; w > 0; w >>= 1) {
    if (t < w) red[t] += red[t + w];
    __syncthreads();
  }
  const float lossSum = red[0];
  __syncthreads();

  float e = 0.f;
  for (int k = t; k < VQ_K; k += 256) {
    float c = 0.f;
#pragma unroll
    for (int h = 0; h < VQ_H; ++h) c += counts[h * VQ_K + k];
    const float p = c * (1.0f / (float)VQ_NH);
    e += p * logf(p + 1e-10f);
  }
  red[t] = e;
  __syncthreads();
  for (int w = 128; w > 0; w >>= 1) {
    if (t < w) red[t] += red[t + w];
    __syncthreads();
  }
  if (t == 0) {
    out[0] = 0.25f * lossSum / (float)VQ_QELEMS;   // COMMITMENT_COST * mean
    out[1 + (size_t)VQ_QELEMS] = expf(-red[0]);    // perplexity
  }
}

// ---------------------------------------------------------------------------
// Host launcher
// ---------------------------------------------------------------------------
extern "C" void kernel_launch(void* const* d_in, const int* in_sizes, int n_in,
                              void* d_out, int out_size, void* d_ws, size_t ws_size,
                              hipStream_t stream) {
  const float* x   = (const float*)d_in[0];   // (16,512,2048) fp32
  const float* emb = (const float*)d_in[1];   // (8,1024,256)  fp32
  float* out = (float*)d_out;                 // [loss | quant(16.7M) | perp | enc(67.1M)]

  char* ws = (char*)d_ws;
  _Float16* embF16      = (_Float16*)(ws);                 //  4 MB   (H*K*C f16)
  float*    normE       = (float*)(ws + 4194304);          // 32 KB   (H*K)
  int*      idxBuf      = (int*)(ws + 4227072);            // 256 KB  (N*H)
  float*    counts      = (float*)(ws + 4489216);          // 32 KB   (H*K)
  float*    lossPartial = (float*)(ws + 4521984);          // 256 KB  (N*H)

  // counts are accumulated with atomics -> must be zeroed every call
  (void)hipMemsetAsync(counts, 0, VQ_H * VQ_K * sizeof(float), stream);

  vq_prep<<<dim3(VQ_H * VQ_K), dim3(256), 0, stream>>>(emb, embF16, normE);

  vq_argmin<<<dim3(VQ_N / 128, VQ_H), dim3(256), 0, stream>>>(x, embF16, normE, idxBuf);

  vq_scatter<<<dim3(VQ_N, VQ_H), dim3(256), 0, stream>>>(
      x, emb, idxBuf,
      out + 1,                              // quantized_st
      out + 1 + (size_t)VQ_QELEMS + 1,      // encodings (offset 16777218)
      counts, lossPartial);

  vq_finalize<<<dim3(1), dim3(256), 0, stream>>>(counts, lossPartial, out);
}
